// PUF_GNN_68444598829509
// MI455X (gfx1250) — matrix operand order repaired and checked by reading, not compiled, not run on hardware
//
#include <hip/hip_runtime.h>
#include <hip/hip_bf16.h>
#include <math.h>

// ---------------------------------------------------------------------------
// Types for CDNA5 WMMA (wave32): v_wmma_f32_16x16x32_bf16
// ---------------------------------------------------------------------------
typedef __attribute__((ext_vector_type(16))) __bf16 v16bf;
typedef __attribute__((ext_vector_type(8)))  __bf16 bf16x8;
typedef __attribute__((ext_vector_type(8)))  float  v8f;

static __device__ inline v16bf bf16_combine(bf16x8 lo, bf16x8 hi) {
    return __builtin_shufflevector(lo, hi, 0,1,2,3,4,5,6,7,8,9,10,11,12,13,14,15);
}

// ---------------------------------------------------------------------------
// CDNA5 async global->LDS DMA (ASYNCcnt-tracked), per CDNA5 ISA
// LDS operand is the wave-relative LDS byte offset = low 32 bits of the generic
// pointer to a __shared__ object (LDS aperture keeps the offset in addr[31:0]).
// ---------------------------------------------------------------------------
static __device__ inline unsigned lds_off(const void* p) {
    return (unsigned)(uintptr_t)p;
}
static __device__ inline void async_copy_b128(unsigned lds_addr, const void* gaddr) {
    asm volatile("global_load_async_to_lds_b128 %0, %1, off"
                 :: "v"(lds_addr), "v"(gaddr) : "memory");
}
static __device__ inline void async_wait0() {
    asm volatile("s_wait_asynccnt 0x0" ::: "memory");
}

// ---------------------------------------------------------------------------
// Problem constants
// ---------------------------------------------------------------------------
#define NN      10000
#define NE      160000
#define ET      170000      // edges + self loops
#define KP1     4128        // 4098 padded to multiple of 32

// ---------------------------------------------------------------------------
// GEMM: C[M,N] = A[M,KP] (bf16, row major) x Bt[N,KP] (bf16, pre-transposed)
// Block = 256 threads = 8 waves; block tile 128x128, k-tile 32, double-buffered
// async-DMA LDS staging. Wave (wm 0..1, wn 0..3) owns a 64x32 tile = 4x2 WMMAs.
// ---------------------------------------------------------------------------
#define AS_STRIDE 48   // 32 K elems + pad; row stride 96B keeps 16B alignment

static __device__ inline void issue_tile(const __bf16* __restrict__ A,
                                         const __bf16* __restrict__ Bt,
                                         __bf16* AsB, __bf16* BsB,
                                         int m0, int n0, int k0, int M, int KP, int tid)
{
    // 128 rows x 32 K per operand = 512 chunks of 8 bf16 (16B); 2 A + 2 B per thread
#pragma unroll
    for (int i = 0; i < 2; ++i) {
        int c  = tid + (i << 8);
        int r  = c >> 2;
        int kk = (c & 3) << 3;
        // A rows >= M feed only discarded C rows: skip issue, stale LDS is harmless
        if (m0 + r < M)
            async_copy_b128(lds_off(AsB + r * AS_STRIDE + kk),
                            A + (size_t)(m0 + r) * KP + k0 + kk);
        async_copy_b128(lds_off(BsB + r * AS_STRIDE + kk),
                        Bt + (size_t)(n0 + r) * KP + k0 + kk);
    }
}

__launch_bounds__(256, 2)
__global__ void gemm_bf16_wmma(const __bf16* __restrict__ A,
                               const __bf16* __restrict__ Bt,
                               float* __restrict__ C,
                               int M, int Ncols, int KP)
{
    __shared__ __bf16 As[2][128 * AS_STRIDE];
    __shared__ __bf16 Bs[2][128 * AS_STRIDE];

    const int tid  = threadIdx.x;
    const int lane = tid & 31;
    const int wave = tid >> 5;
    const int wm   = wave & 1;      // 2 waves along M
    const int wn   = wave >> 1;     // 4 waves along N
    const int m0   = blockIdx.y * 128;
    const int n0   = blockIdx.x * 128;

    v8f acc[4][2];
#pragma unroll
    for (int t = 0; t < 4; ++t)
#pragma unroll
        for (int u = 0; u < 2; ++u)
#pragma unroll
            for (int r = 0; r < 8; ++r) acc[t][u][r] = 0.0f;

    const int nkt = KP >> 5;
    // prologue: tile 0 in flight
    issue_tile(A, Bt, As[0], Bs[0], m0, n0, 0, M, KP, tid);

    for (int kt = 0; kt < nkt; ++kt) {
        const int cur = kt & 1;
        async_wait0();        // my tile-kt chunks resident in LDS
        __syncthreads();      // everyone's chunks resident; prev compute done (WAR safe)
        if (kt + 1 < nkt)     // overlap tile kt+1 DMA with tile kt WMMAs
            issue_tile(A, Bt, As[cur ^ 1], Bs[cur ^ 1], m0, n0, (kt + 1) << 5, M, KP, tid);

        const __bf16* Asb = As[cur];
        const __bf16* Bsb = Bs[cur];

        // ---- A fragments: 16-bit A 16x32 layout
        //      lanes 0-15 : K=0..7 in v0..3, K=16..23 in v4..7 (row M=lane)
        //      lanes16-31 : K=8..15 / K=24..31
        v16bf af[4], bfv[2];
        {
            const int kb = (lane >> 4) << 3;  // 0 or 8
#pragma unroll
            for (int t = 0; t < 4; ++t) {
                const __bf16* p = Asb + (wm * 64 + t * 16 + (lane & 15)) * AS_STRIDE + kb;
                bf16x8 lo = *(const bf16x8*)p;         // K = kb..kb+7
                bf16x8 hi = *(const bf16x8*)(p + 16);  // K = 16+kb..16+kb+7
                af[t] = bf16_combine(lo, hi);
            }
            // ---- B fragments: 16-bit B 32x16 layout
            //      lanes 0-15 hold K=0..7,16..23, lanes 16-31 hold K=8..15,24..31 (col N=lane)
            const int kb2 = (lane >> 4) << 3; // 0 or 8
#pragma unroll
            for (int u = 0; u < 2; ++u) {
                const __bf16* p = Bsb + (wn * 32 + u * 16 + (lane & 15)) * AS_STRIDE + kb2;
                bf16x8 lo = *(const bf16x8*)p;
                bf16x8 hi = *(const bf16x8*)(p + 16);
                bfv[u] = bf16_combine(lo, hi);
            }
        }
#pragma unroll
        for (int t = 0; t < 4; ++t)
#pragma unroll
            for (int u = 0; u < 2; ++u)
                acc[t][u] = __builtin_amdgcn_wmma_f32_16x16x32_bf16(
                    false, af[t], false, bfv[u], (short)0, acc[t][u], false, false);
    }

    // ---- store D: VGPR r holds M=r (lanes 0-15) / M=8+r (lanes 16-31), N=lane&15
#pragma unroll
    for (int t = 0; t < 4; ++t) {
        const int rowb = m0 + wm * 64 + t * 16 + ((lane >> 4) << 3);
#pragma unroll
        for (int u = 0; u < 2; ++u) {
            const int col = n0 + wn * 32 + u * 16 + (lane & 15);
#pragma unroll
            for (int r = 0; r < 8; ++r) {
                const int row = rowb + r;
                if (row < M) C[(size_t)row * Ncols + col] = acc[t][u][r];
            }
        }
    }
}

// ---------------------------------------------------------------------------
// fp32 -> bf16 conversions (with K zero-padding); weights also transposed
// ---------------------------------------------------------------------------
__global__ void convert_pad_kernel(const float* __restrict__ in, __bf16* __restrict__ out,
                                   int rows, int K, int KP)
{
    size_t i = (size_t)blockIdx.x * blockDim.x + threadIdx.x;
    if (i >= (size_t)rows * KP) return;
    int r = (int)(i / KP), k = (int)(i % KP);
    out[i] = (__bf16)(k < K ? in[(size_t)r * K + k] : 0.0f);
}

__global__ void convert_wt_kernel(const float* __restrict__ W, __bf16* __restrict__ out,
                                  int K, int Ncols, int KP)
{
    // out[n][k] = W[k][n], zero-padded in K
    size_t i = (size_t)blockIdx.x * blockDim.x + threadIdx.x;
    if (i >= (size_t)Ncols * KP) return;
    int n = (int)(i / KP), k = (int)(i % KP);
    out[i] = (__bf16)(k < K ? W[(size_t)k * Ncols + n] : 0.0f);
}

__global__ void fill_kernel(float* __restrict__ p, size_t count, float v)
{
    size_t i = (size_t)blockIdx.x * blockDim.x + threadIdx.x;
    if (i < count) p[i] = v;
}

// ---------------------------------------------------------------------------
// Per-(node,head) attention coefficients: as[n,h] = xw[n,h,:].a_s[h,:], etc.
// ---------------------------------------------------------------------------
__global__ void alpha_reduce_kernel(const float* __restrict__ xw,
                                    const float* __restrict__ a_s,
                                    const float* __restrict__ a_d,
                                    float* __restrict__ as_out,
                                    float* __restrict__ ad_out, int H)
{
    const int n = blockIdx.x, h = blockIdx.y, tid = threadIdx.x; // 128 threads
    const float* row = xw + ((size_t)n * H + h) * 512;
    const float* vs  = a_s + h * 512;
    const float* vd  = a_d + h * 512;
    float ss = 0.0f, sd = 0.0f;
    for (int c = tid; c < 512; c += 128) {
        float v = row[c];
        ss += v * vs[c];
        sd += v * vd[c];
    }
    __shared__ float sS[128], sD[128];
    sS[tid] = ss; sD[tid] = sd;
    __syncthreads();
    for (int s = 64; s > 0; s >>= 1) {
        if (tid < s) { sS[tid] += sS[tid + s]; sD[tid] += sD[tid + s]; }
        __syncthreads();
    }
    if (tid == 0) { as_out[n * H + h] = sS[0]; ad_out[n * H + h] = sD[0]; }
}

// ---------------------------------------------------------------------------
// Edge phases (segment softmax over dst + scatter)
// ---------------------------------------------------------------------------
static __device__ inline void atomicMaxF(float* addr, float val) {
    if (val >= 0.0f) atomicMax((int*)addr, __float_as_int(val));
    else             atomicMin((unsigned int*)addr, (unsigned int)__float_as_int(val));
}

static __device__ inline void edge_endpoints(const int* ei, int e, int& s, int& d) {
    if (e < NE) { s = ei[e]; d = ei[NE + e]; }
    else        { s = d = e - NE; }   // self loops
}

__global__ void edge_max_kernel(const int* __restrict__ ei,
                                const float* __restrict__ as_, const float* __restrict__ ad_,
                                float* __restrict__ nmax, int H)
{
    int e = blockIdx.x * blockDim.x + threadIdx.x;
    if (e >= ET) return;
    int s, d; edge_endpoints(ei, e, s, d);
    for (int h = 0; h < H; ++h) {
        float x = as_[s * H + h] + ad_[d * H + h];
        x = x > 0.0f ? x : 0.2f * x;              // leaky_relu(0.2)
        atomicMaxF(&nmax[d * H + h], x);
    }
}

__global__ void edge_expsum_kernel(const int* __restrict__ ei,
                                   const float* __restrict__ as_, const float* __restrict__ ad_,
                                   const float* __restrict__ nmax,
                                   float* __restrict__ nsum, float* __restrict__ alpha_e, int H)
{
    int e = blockIdx.x * blockDim.x + threadIdx.x;
    if (e >= ET) return;
    int s, d; edge_endpoints(ei, e, s, d);
    for (int h = 0; h < H; ++h) {
        float x = as_[s * H + h] + ad_[d * H + h];
        x = x > 0.0f ? x : 0.2f * x;
        float ex = __expf(x - nmax[d * H + h]);
        alpha_e[(size_t)e * H + h] = ex;
        atomicAdd(&nsum[d * H + h], ex);
    }
}

__global__ void edge_scatter_kernel(const int* __restrict__ ei,
                                    const float* __restrict__ alpha_e,
                                    const float* __restrict__ nsum,
                                    const float* __restrict__ xw,
                                    float* __restrict__ agg, int H)
{
    const int e = blockIdx.x;
    int s, d; edge_endpoints(ei, e, s, d);
    __shared__ float coef[4];
    if (threadIdx.x < H)
        coef[threadIdx.x] = alpha_e[(size_t)e * H + threadIdx.x] /
                            (nsum[d * H + threadIdx.x] + 1e-16f);
    __syncthreads();
    const int HC = H << 9;
    const float* src = xw  + (size_t)s * HC;
    float*       dst = agg + (size_t)d * HC;
    for (int c = threadIdx.x; c < HC; c += blockDim.x)
        atomicAdd(&dst[c], coef[c >> 9] * src[c]);
}

// bias + ReLU, emit bf16 activations for next layer's GEMM
__global__ void finalize_kernel(const float* __restrict__ agg, const float* __restrict__ bias,
                                __bf16* __restrict__ outbf, int total, int HC)
{
    int i = blockIdx.x * blockDim.x + threadIdx.x;
    if (i >= total) return;
    float v = agg[i] + bias[i % HC];
    v = v > 0.0f ? v : 0.0f;
    outbf[i] = (__bf16)v;
}

// ---------------------------------------------------------------------------
// Final heads: h3 = agg3 + b3 (no relu); group = h3@Wg + bg, nongroup = h3@Wn + bn
// One wave per node, shuffle reduction (wave32).
// ---------------------------------------------------------------------------
__global__ void logits_kernel(const float* __restrict__ agg3, const float* __restrict__ b3,
                              const float* __restrict__ Wg, const float* __restrict__ bg,
                              const float* __restrict__ Wn, const float* __restrict__ bn,
                              float* __restrict__ out)
{
    const int wave = threadIdx.x >> 5, lane = threadIdx.x & 31;
    const int n = blockIdx.x * 8 + wave;
    if (n >= NN) return;
    const float* row = agg3 + (size_t)n * 512;
    float g0 = 0, g1 = 0, g2 = 0, gn = 0;
    for (int c = lane; c < 512; c += 32) {
        float t = row[c] + b3[c];
        g0 += t * Wg[c * 3 + 0];
        g1 += t * Wg[c * 3 + 1];
        g2 += t * Wg[c * 3 + 2];
        gn += t * Wn[c];
    }
#pragma unroll
    for (int off = 16; off; off >>= 1) {
        g0 += __shfl_xor(g0, off, 32);
        g1 += __shfl_xor(g1, off, 32);
        g2 += __shfl_xor(g2, off, 32);
        gn += __shfl_xor(gn, off, 32);
    }
    if (lane == 0) {
        out[n * 3 + 0] = g0 + bg[0];
        out[n * 3 + 1] = g1 + bg[1];
        out[n * 3 + 2] = g2 + bg[2];
        out[NN * 3 + n] = gn + bn[0];
    }
}

// ---------------------------------------------------------------------------
// Host orchestration
// ---------------------------------------------------------------------------
static inline unsigned cdiv(size_t a, unsigned b) { return (unsigned)((a + b - 1) / b); }

extern "C" void kernel_launch(void* const* d_in, const int* in_sizes, int n_in,
                              void* d_out, int out_size, void* d_ws, size_t ws_size,
                              hipStream_t stream)
{
    const float* x   = (const float*)d_in[0];
    const int*   ei  = (const int*)d_in[1];
    const float* W1  = (const float*)d_in[2];
    const float* a1s = (const float*)d_in[3];
    const float* a1d = (const float*)d_in[4];
    const float* b1  = (const float*)d_in[5];
    const float* W2  = (const float*)d_in[6];
    const float* a2s = (const float*)d_in[7];
    const float* a2d = (const float*)d_in[8];
    const float* b2  = (const float*)d_in[9];
    const float* W3  = (const float*)d_in[10];
    const float* a3s = (const float*)d_in[11];
    const float* a3d = (const float*)d_in[12];
    const float* b3  = (const float*)d_in[13];
    const float* Wg  = (const float*)d_in[14];
    const float* bg  = (const float*)d_in[15];
    const float* Wn  = (const float*)d_in[16];
    const float* bn  = (const float*)d_in[17];
    float* out = (float*)d_out;

    // ---- workspace carve (bump allocator, 256B aligned), with region reuse
    size_t off = 0;
    char* base = (char*)d_ws;
    auto alloc = [&](size_t bytes) -> void* {
        off = (off + 255) & ~(size_t)255;
        void* p = base + off;
        off += bytes;
        return p;
    };
    __bf16* xbf    = (__bf16*)alloc((size_t)NN * KP1 * 2);      // later reused as agg1 (f32)
    __bf16* W1t    = (__bf16*)alloc((size_t)2048 * KP1 * 2);
    float*  xw1    = (float*) alloc((size_t)NN * 2048 * 4);     // later: xw2 + agg2
    __bf16* h1bf   = (__bf16*)alloc((size_t)NN * 2048 * 2);     // later: h2bf + xw3
    __bf16* W2t    = (__bf16*)alloc((size_t)1024 * 2048 * 2);
    __bf16* W3t    = (__bf16*)alloc((size_t)512 * 1024 * 2);
    float*  asb    = (float*) alloc((size_t)NN * 4 * 4);
    float*  adb    = (float*) alloc((size_t)NN * 4 * 4);
    float*  nmax   = (float*) alloc((size_t)NN * 4 * 4);
    float*  nsum   = (float*) alloc((size_t)NN * 4 * 4);
    float*  alphae = (float*) alloc((size_t)ET * 4 * 4);
    float*  agg3   = (float*) alloc((size_t)NN * 512 * 4);
    // region reuse (all uses stream-ordered after producers are dead):
    float*  agg1 = (float*)xbf;                                  // 81.92MB <= 82.56MB
    float*  xw2  = xw1;
    float*  agg2 = xw1 + (size_t)NN * 1024;
    __bf16* h2bf = h1bf;
    float*  xw3  = (float*)(h1bf + (size_t)NN * 1024);

    const unsigned T = 256;

    // ---- one-time conversions (run every call; deterministic)
    convert_pad_kernel<<<cdiv((size_t)NN * KP1, T), T, 0, stream>>>(x, xbf, NN, 4098, KP1);
    convert_wt_kernel<<<cdiv((size_t)2048 * KP1, T), T, 0, stream>>>(W1, W1t, 4098, 2048, KP1);
    convert_wt_kernel<<<cdiv((size_t)1024 * 2048, T), T, 0, stream>>>(W2, W2t, 2048, 1024, 2048);
    convert_wt_kernel<<<cdiv((size_t)512 * 1024, T), T, 0, stream>>>(W3, W3t, 1024, 512, 1024);

    const unsigned gy = (NN + 127) / 128; // 79
    const unsigned ge = cdiv(ET, T);

    // ================= Layer 1: H=4, HC=2048 =================
    gemm_bf16_wmma<<<dim3(2048 / 128, gy), T, 0, stream>>>(xbf, W1t, xw1, NN, 2048, KP1);
    alpha_reduce_kernel<<<dim3(NN, 4), 128, 0, stream>>>(xw1, a1s, a1d, asb, adb, 4);
    fill_kernel<<<cdiv((size_t)NN * 4, T), T, 0, stream>>>(nmax, (size_t)NN * 4, -__builtin_inff());
    fill_kernel<<<cdiv((size_t)NN * 4, T), T, 0, stream>>>(nsum, (size_t)NN * 4, 0.0f);
    fill_kernel<<<cdiv((size_t)NN * 2048, T), T, 0, stream>>>(agg1, (size_t)NN * 2048, 0.0f);
    edge_max_kernel<<<ge, T, 0, stream>>>(ei, asb, adb, nmax, 4);
    edge_expsum_kernel<<<ge, T, 0, stream>>>(ei, asb, adb, nmax, nsum, alphae, 4);
    edge_scatter_kernel<<<ET, T, 0, stream>>>(ei, alphae, nsum, xw1, agg1, 4);
    finalize_kernel<<<cdiv((size_t)NN * 2048, T), T, 0, stream>>>(agg1, b1, h1bf, NN * 2048, 2048);

    // ================= Layer 2: H=2, HC=1024 =================
    gemm_bf16_wmma<<<dim3(1024 / 128, gy), T, 0, stream>>>(h1bf, W2t, xw2, NN, 1024, 2048);
    alpha_reduce_kernel<<<dim3(NN, 2), 128, 0, stream>>>(xw2, a2s, a2d, asb, adb, 2);
    fill_kernel<<<cdiv((size_t)NN * 2, T), T, 0, stream>>>(nmax, (size_t)NN * 2, -__builtin_inff());
    fill_kernel<<<cdiv((size_t)NN * 2, T), T, 0, stream>>>(nsum, (size_t)NN * 2, 0.0f);
    fill_kernel<<<cdiv((size_t)NN * 1024, T), T, 0, stream>>>(agg2, (size_t)NN * 1024, 0.0f);
    edge_max_kernel<<<ge, T, 0, stream>>>(ei, asb, adb, nmax, 2);
    edge_expsum_kernel<<<ge, T, 0, stream>>>(ei, asb, adb, nmax, nsum, alphae, 2);
    edge_scatter_kernel<<<ET, T, 0, stream>>>(ei, alphae, nsum, xw2, agg2, 2);
    finalize_kernel<<<cdiv((size_t)NN * 1024, T), T, 0, stream>>>(agg2, b2, h2bf, NN * 1024, 1024);

    // ================= Layer 3: H=1, HC=512 =================
    gemm_bf16_wmma<<<dim3(512 / 128, gy), T, 0, stream>>>(h2bf, W3t, xw3, NN, 512, 1024);
    alpha_reduce_kernel<<<dim3(NN, 1), 128, 0, stream>>>(xw3, a3s, a3d, asb, adb, 1);
    fill_kernel<<<cdiv((size_t)NN, T), T, 0, stream>>>(nmax, (size_t)NN, -__builtin_inff());
    fill_kernel<<<cdiv((size_t)NN, T), T, 0, stream>>>(nsum, (size_t)NN, 0.0f);
    fill_kernel<<<cdiv((size_t)NN * 512, T), T, 0, stream>>>(agg3, (size_t)NN * 512, 0.0f);
    edge_max_kernel<<<ge, T, 0, stream>>>(ei, asb, adb, nmax, 1);
    edge_expsum_kernel<<<ge, T, 0, stream>>>(ei, asb, adb, nmax, nsum, alphae, 1);
    edge_scatter_kernel<<<ET, T, 0, stream>>>(ei, alphae, nsum, xw3, agg3, 1);

    // ================= Output heads (b3 folded in) =================
    logits_kernel<<<cdiv((size_t)NN, 8), T, 0, stream>>>(agg3, b3, Wg, bg, Wn, bn, out);
}
